// PriceActionAttention_36043365548576
// MI455X (gfx1250) — matrix operand order repaired and checked by reading, not compiled, not run on hardware
//
#include <hip/hip_runtime.h>

#define D_MODEL   1024
#define N_HEADS   16
#define HEAD_DIM  64
#define BATCH     2
#define SEQ       2048
#define BS        (BATCH * SEQ)   // 4096 rows

typedef __bf16 bf16;
typedef __attribute__((ext_vector_type(16))) __bf16 v16bf;
typedef __attribute__((ext_vector_type(8)))  __bf16 v8bf;
typedef __attribute__((ext_vector_type(8)))  float  v8f;
typedef __attribute__((ext_vector_type(4)))  int    v4i;
typedef __attribute__((ext_vector_type(4)))  unsigned int v4u;

#define AS1 __attribute__((address_space(1)))
#define AS3 __attribute__((address_space(3)))

#if defined(__has_builtin)
#  if __has_builtin(__builtin_amdgcn_global_load_async_to_lds_b128)
#    define HAVE_ASYNC_LDS 1
#  endif
#endif

// ---------------------------------------------------------------------------
// WMMA helpers
// ---------------------------------------------------------------------------
__device__ __forceinline__ v8f wmma_bf16(v16bf a, v16bf b, v8f c) {
  return __builtin_amdgcn_wmma_f32_16x16x32_bf16(
      /*neg_a=*/false, a, /*neg_b=*/false, b,
      /*c_mod=*/(short)0, c, /*reuse_a=*/false, /*reuse_b=*/false);
}

// A-fragment (16x32, bf16): lane holds row M = lane%16; K chunks are
// [half*8, half*8+8) and [16+half*8, 16+half*8+8)  (ISA 7.12.2 table).
__device__ __forceinline__ v16bf load_A_frag(const bf16* rowk, int half) {
  v8bf lo = *(const v8bf*)(rowk + half * 8);
  v8bf hi = *(const v8bf*)(rowk + 16 + half * 8);
  return __builtin_shufflevector(lo, hi, 0, 1, 2, 3, 4, 5, 6, 7,
                                 8, 9, 10, 11, 12, 13, 14, 15);
}

// B-fragment (32x16, bf16): lane holds column N = lane%16; K chunks are
// [half*8, half*8+8) and [16+half*8, 16+half*8+8) (same as A).
__device__ __forceinline__ v16bf load_B_frag(const bf16* colk, int half) {
  v8bf lo = *(const v8bf*)(colk + half * 8);
  v8bf hi = *(const v8bf*)(colk + 16 + half * 8);
  return __builtin_shufflevector(lo, hi, 0, 1, 2, 3, 4, 5, 6, 7,
                                 8, 9, 10, 11, 12, 13, 14, 15);
}

// ---------------------------------------------------------------------------
// 16-byte global -> LDS staging (async DMA, ASYNCcnt-tracked)
// ---------------------------------------------------------------------------
__device__ __forceinline__ void copy16_g2l(bf16* ldst, const bf16* gsrc) {
#ifdef HAVE_ASYNC_LDS
  __builtin_amdgcn_global_load_async_to_lds_b128(
      (AS1 v4i*)gsrc, (AS3 v4i*)ldst, /*offset=*/0, /*cpol=*/0);
#else
  *(v4u*)ldst = *(const v4u*)gsrc;
#endif
}

__device__ __forceinline__ void staging_wait() {
#ifdef HAVE_ASYNC_LDS
#  if __has_builtin(__builtin_amdgcn_s_wait_asynccnt)
  __builtin_amdgcn_s_wait_asynccnt(0);
#  else
  asm volatile("s_wait_asynccnt 0x0" ::: "memory");
#  endif
#endif
}

// ---------------------------------------------------------------------------
// fp32 -> bf16 conversion (grid-stride)
// ---------------------------------------------------------------------------
__global__ void f32_to_bf16_kernel(const float* __restrict__ in,
                                   bf16* __restrict__ out, int n) {
  int i = blockIdx.x * blockDim.x + threadIdx.x;
  int stride = gridDim.x * blockDim.x;
  for (; i < n; i += stride) out[i] = (bf16)in[i];
}

// ---------------------------------------------------------------------------
// softmax over the 16 per-head pattern weights
// ---------------------------------------------------------------------------
__global__ void pattern_softmax_kernel(const float* __restrict__ pw,
                                       float* __restrict__ pat) {
  if (threadIdx.x == 0) {
    float m = -1e30f;
    for (int h = 0; h < N_HEADS; ++h) m = fmaxf(m, pw[h]);
    float e[N_HEADS];
    float s = 0.f;
    for (int h = 0; h < N_HEADS; ++h) { e[h] = __expf(pw[h] - m); s += e[h]; }
    float inv = 1.0f / s;
    for (int h = 0; h < N_HEADS; ++h) pat[h] = e[h] * inv;
  }
}

// ---------------------------------------------------------------------------
// GEMM: out[M=4096, N=1024] = A[4096,1024] @ W[N,K=1024]^T + bias
// Double-buffered async LDS staging of 128x32 A/B slabs; DMA overlaps WMMA.
// MODE 0: bf16 row-major (Q,K)  MODE 1: bf16 Vt[(b*1024+e)][s]  MODE 2: fp32
// WG: 256 threads = 8 waves; wave tile 64(M)x32(N); WG tile 128x128.
// ---------------------------------------------------------------------------
template <int MODE>
__global__ __launch_bounds__(256) void gemm_bf16_kernel(
    const bf16* __restrict__ A, const bf16* __restrict__ W,
    const float* __restrict__ bias, void* __restrict__ outp) {
  __shared__ __align__(128) bf16 Asl[2][128][32];  // 16 KB
  __shared__ __align__(128) bf16 Bsl[2][128][32];  // 16 KB

  const int tid  = threadIdx.x;
  const int lane = tid & 31;
  const int wid  = tid >> 5;
  const int half = lane >> 4;
  const int l16  = lane & 15;

  const int m0 = blockIdx.x * 128;                 // WG tile origin
  const int n0 = blockIdx.y * 128;
  const int wm = (wid & 1) * 64;                   // wave offsets in WG tile
  const int wn = (wid >> 1) * 32;

  // staging map: slab = 128 rows x 64 B = 512 x 16 B chunks; 2 per thread
  const int s_row = tid >> 1;                      // 0..127

  auto stage = [&](int buf, int k) {
#pragma unroll
    for (int c = 0; c < 2; ++c) {
      const int e = ((tid & 1) * 2 + c) * 8;       // bf16 offset in 32-row
      copy16_g2l(&Asl[buf][s_row][e], A + (size_t)(m0 + s_row) * D_MODEL + k + e);
      copy16_g2l(&Bsl[buf][s_row][e], W + (size_t)(n0 + s_row) * D_MODEL + k + e);
    }
  };

  const v8f vzero = {0.f, 0.f, 0.f, 0.f, 0.f, 0.f, 0.f, 0.f};
  v8f acc[4][2];
#pragma unroll
  for (int i = 0; i < 4; ++i)
#pragma unroll
    for (int j = 0; j < 2; ++j) acc[i][j] = vzero;

  stage(0, 0);
  staging_wait();
  __syncthreads();

  int buf = 0;
  for (int k = 0; k < D_MODEL; k += 32, buf ^= 1) {
    if (k + 32 < D_MODEL) stage(buf ^ 1, k + 32);  // DMA next slab now

    v16bf a[4], b[2];
#pragma unroll
    for (int i = 0; i < 4; ++i)
      a[i] = load_A_frag(&Asl[buf][wm + i * 16 + l16][0], half);
#pragma unroll
    for (int j = 0; j < 2; ++j)
      b[j] = load_B_frag(&Bsl[buf][wn + j * 16 + l16][0], half);

#pragma unroll
    for (int i = 0; i < 4; ++i)
#pragma unroll
      for (int j = 0; j < 2; ++j)
        acc[i][j] = wmma_bf16(a[i], b[j], acc[i][j]);

    staging_wait();    // next-slab DMA landed
    __syncthreads();   // everyone done reading current slab
  }

  // C layout: vgpr r -> row (r + 8*half), lane -> col l16
#pragma unroll
  for (int i = 0; i < 4; ++i) {
#pragma unroll
    for (int j = 0; j < 2; ++j) {
#pragma unroll
      for (int r = 0; r < 8; ++r) {
        const int M = m0 + wm + i * 16 + r + half * 8;
        const int N = n0 + wn + j * 16 + l16;
        const float v = acc[i][j][r] + bias[N];
        if (MODE == 0) {
          ((bf16*)outp)[(size_t)M * D_MODEL + N] = (bf16)v;
        } else if (MODE == 1) {
          const int bb = M >> 11;
          const int s  = M & (SEQ - 1);
          ((bf16*)outp)[((size_t)(bb * D_MODEL + N)) * SEQ + s] = (bf16)v;
        } else {
          ((float*)outp)[(size_t)M * D_MODEL + N] = v;
        }
      }
    }
  }
}

// ---------------------------------------------------------------------------
// Flash attention, 64-key blocks; K/V tiles double-buffered via async DMA.
// grid = (S/128, B*H); block = 256 (8 waves, each owns 16 query rows).
// ---------------------------------------------------------------------------
__global__ __launch_bounds__(256) void flash_attn_kernel(
    const bf16* __restrict__ Q,   // [4096,1024]
    const bf16* __restrict__ K,   // [4096,1024]
    const bf16* __restrict__ Vt,  // [(b*1024+e)][2048]
    const float* __restrict__ pat,
    bf16* __restrict__ ctx) {     // [4096,1024]
  __shared__ __align__(128) bf16 Ktile[2][64][64];   // [t_local][hd]  16 KB
  __shared__ __align__(128) bf16 Vtile[2][64][64];   // [hd][t_local]  16 KB
  __shared__ __align__(128) bf16 Plds[8][16][64];    // per-wave P     16 KB

  const int tid  = threadIdx.x;
  const int lane = tid & 31;
  const int wid  = tid >> 5;
  const int half = lane >> 4;
  const int l16  = lane & 15;

  const int bh = blockIdx.y;
  const int b  = bh >> 4;
  const int h  = bh & 15;
  const int m0 = blockIdx.x * 128 + wid * 16;      // query row block
  const size_t qrow0 = (size_t)(b * SEQ + m0);
  const int colq = h * HEAD_DIM;
  const float scale = pat[h] * 0.125f;             // softmax(pw)[h]/sqrt(64)

  // Q fragments (A layout), K-dim = head_dim = 64 -> two k-steps of 32
  v16bf qf[2];
#pragma unroll
  for (int kk = 0; kk < 2; ++kk)
    qf[kk] = load_A_frag(Q + (qrow0 + l16) * D_MODEL + colq + kk * 32, half);

  const v8f vzero = {0.f, 0.f, 0.f, 0.f, 0.f, 0.f, 0.f, 0.f};
  float mi[8], li[8];
  v8f oacc[4];
#pragma unroll
  for (int r = 0; r < 8; ++r) { mi[r] = -1e30f; li[r] = 0.f; }
#pragma unroll
  for (int n = 0; n < 4; ++n) oacc[n] = vzero;

  // staging map: tile = 64 rows x 128 B; 512 16-byte chunks; 2 per thread
  const int c_row = tid >> 2;          // 0..63
  const int c_col = (tid & 3) * 16;    // bf16 element offset (0,16,32,48)

  auto stageKV = [&](int buf, int tb) {
#pragma unroll
    for (int c = 0; c < 2; ++c) {
      const int e = c_col + c * 8;     // 8 bf16 = 16 bytes per chunk
      copy16_g2l(&Ktile[buf][c_row][e],
                 K + (size_t)(b * SEQ + tb + c_row) * D_MODEL + colq + e);
      copy16_g2l(&Vtile[buf][c_row][e],
                 Vt + (size_t)(b * D_MODEL + colq + c_row) * SEQ + tb + e);
    }
  };

  stageKV(0, 0);
  staging_wait();
  __syncthreads();

  int buf = 0;
  for (int tb = 0; tb < SEQ; tb += 64, buf ^= 1) {
    if (tb + 64 < SEQ) stageKV(buf ^ 1, tb + 64);  // DMA next tiles now

    // ---- scores: four 16x16 tiles over keys [tb, tb+64) ----
    v8f s[4] = {vzero, vzero, vzero, vzero};
#pragma unroll
    for (int j = 0; j < 4; ++j)
#pragma unroll
      for (int kk = 0; kk < 2; ++kk) {
        v16bf kb = load_B_frag(&Ktile[buf][j * 16 + l16][kk * 32], half);
        s[j] = wmma_bf16(qf[kk], kb, s[j]);
      }

    // ---- online softmax: one butterfly per stat per 64 keys ----
    float alpha[8];
#pragma unroll
    for (int r = 0; r < 8; ++r) {
      float s0 = s[0][r] * scale, s1 = s[1][r] * scale;
      float s2 = s[2][r] * scale, s3 = s[3][r] * scale;
      float v = fmaxf(fmaxf(s0, s1), fmaxf(s2, s3));
      v = fmaxf(v, __shfl_xor(v, 1, 16));
      v = fmaxf(v, __shfl_xor(v, 2, 16));
      v = fmaxf(v, __shfl_xor(v, 4, 16));
      v = fmaxf(v, __shfl_xor(v, 8, 16));
      const float mnew = fmaxf(mi[r], v);
      alpha[r] = __expf(mi[r] - mnew);
      const float p0 = __expf(s0 - mnew);
      const float p1 = __expf(s1 - mnew);
      const float p2 = __expf(s2 - mnew);
      const float p3 = __expf(s3 - mnew);
      s[0][r] = p0; s[1][r] = p1; s[2][r] = p2; s[3][r] = p3;
      float ls = (p0 + p1) + (p2 + p3);
      ls += __shfl_xor(ls, 1, 16);
      ls += __shfl_xor(ls, 2, 16);
      ls += __shfl_xor(ls, 4, 16);
      ls += __shfl_xor(ls, 8, 16);
      li[r] = li[r] * alpha[r] + ls;
      mi[r] = mnew;
    }
#pragma unroll
    for (int n = 0; n < 4; ++n)
#pragma unroll
      for (int r = 0; r < 8; ++r) oacc[n][r] *= alpha[r];

    // ---- transpose P (C layout -> A layout) through per-wave LDS tile ----
#pragma unroll
    for (int j = 0; j < 4; ++j)
#pragma unroll
      for (int r = 0; r < 8; ++r)
        Plds[wid][r + half * 8][j * 16 + l16] = (bf16)s[j][r];
    asm volatile("s_wait_dscnt 0x0" ::: "memory");
    v16bf pA[2];
#pragma unroll
    for (int kk = 0; kk < 2; ++kk)
      pA[kk] = load_A_frag(&Plds[wid][l16][kk * 32], half);

    // ---- context: P(16x64) x V(64x64) ----
#pragma unroll
    for (int n = 0; n < 4; ++n)
#pragma unroll
      for (int kk = 0; kk < 2; ++kk) {
        v16bf vb = load_B_frag(&Vtile[buf][n * 16 + l16][kk * 32], half);
        oacc[n] = wmma_bf16(pA[kk], vb, oacc[n]);
      }

    staging_wait();    // next K/V tiles landed
    __syncthreads();   // everyone done reading current tiles
  }

  // ---- finalize: O / l, store bf16 into context buffer ----
#pragma unroll
  for (int r = 0; r < 8; ++r) li[r] = 1.0f / li[r];
#pragma unroll
  for (int n = 0; n < 4; ++n)
#pragma unroll
    for (int r = 0; r < 8; ++r) {
      const float v = oacc[n][r] * li[r];
      ctx[(qrow0 + r + half * 8) * D_MODEL + colq + n * 16 + l16] = (bf16)v;
    }
}

// ---------------------------------------------------------------------------
// Launch
// ---------------------------------------------------------------------------
extern "C" void kernel_launch(void* const* d_in, const int* in_sizes, int n_in,
                              void* d_out, int out_size, void* d_ws,
                              size_t ws_size, hipStream_t stream) {
  const float* x  = (const float*)d_in[0];
  const float* Wq = (const float*)d_in[1];
  const float* bq = (const float*)d_in[2];
  const float* Wk = (const float*)d_in[3];
  const float* bk = (const float*)d_in[4];
  const float* Wv = (const float*)d_in[5];
  const float* bv = (const float*)d_in[6];
  const float* Wo = (const float*)d_in[7];
  const float* bo = (const float*)d_in[8];
  const float* pw = (const float*)d_in[9];

  char* ws = (char*)d_ws;
  const size_t actB = (size_t)BS * D_MODEL * sizeof(bf16);       // 8 MB
  const size_t wB   = (size_t)D_MODEL * D_MODEL * sizeof(bf16);  // 2 MB
  bf16* xb  = (bf16*)ws;            ws += actB;
  bf16* wqb = (bf16*)ws;            ws += wB;
  bf16* wkb = (bf16*)ws;            ws += wB;
  bf16* wvb = (bf16*)ws;            ws += wB;
  bf16* wob = (bf16*)ws;            ws += wB;
  bf16* Qb  = (bf16*)ws;            ws += actB;
  bf16* Kb  = (bf16*)ws;            ws += actB;
  bf16* Vt  = (bf16*)ws;            ws += actB;
  bf16* ctx = (bf16*)ws;            ws += actB;
  float* pat = (float*)ws;

  const int nAct = BS * D_MODEL;
  const int nW   = D_MODEL * D_MODEL;

  f32_to_bf16_kernel<<<2048, 256, 0, stream>>>(x, xb, nAct);
  f32_to_bf16_kernel<<<1024, 256, 0, stream>>>(Wq, wqb, nW);
  f32_to_bf16_kernel<<<1024, 256, 0, stream>>>(Wk, wkb, nW);
  f32_to_bf16_kernel<<<1024, 256, 0, stream>>>(Wv, wvb, nW);
  f32_to_bf16_kernel<<<1024, 256, 0, stream>>>(Wo, wob, nW);
  pattern_softmax_kernel<<<1, 32, 0, stream>>>(pw, pat);

  dim3 ggrid(BS / 128, D_MODEL / 128);  // (32, 8)
  gemm_bf16_kernel<0><<<ggrid, 256, 0, stream>>>(xb, wqb, bq, (void*)Qb);
  gemm_bf16_kernel<0><<<ggrid, 256, 0, stream>>>(xb, wkb, bk, (void*)Kb);
  gemm_bf16_kernel<1><<<ggrid, 256, 0, stream>>>(xb, wvb, bv, (void*)Vt);

  dim3 agrid(SEQ / 128, BATCH * N_HEADS);  // (16, 32)
  flash_attn_kernel<<<agrid, 256, 0, stream>>>(Qb, Kb, Vt, pat, ctx);

  gemm_bf16_kernel<2><<<ggrid, 256, 0, stream>>>(ctx, wob, bo, d_out);
}
